// MultiHeadAttention_16578573762716
// MI455X (gfx1250) — compile-verified, matrix operand order checked by reading, not run on hardware
//
#include <hip/hip_runtime.h>
#include <hip/hip_bf16.h>

typedef __attribute__((ext_vector_type(16))) _Float16 v16h;
typedef __attribute__((ext_vector_type(8)))  _Float16 v8h;
typedef __attribute__((ext_vector_type(4)))  _Float16 v4h;
typedef __attribute__((ext_vector_type(8)))  float    v8f;

constexpr int BATCH = 4;
constexpr int NSEQ  = 2048;
constexpr int EMB   = 512;
constexpr int HEADS = 8;
constexpr int DHEAD = 64;
constexpr int BN    = BATCH * NSEQ;             // 8192 rows
constexpr float QSCALE = 0.044194173824159216f; // 1/sqrt(512)

// Load an f16 A/B fragment piece for one lane: elements 0..7 from p[0..7],
// elements 8..15 from p[16..23] (caller passes p = rowbase + kbase + g*8,
// matching the CDNA5 16-bit fragment K pattern). Works for global and LDS.
static __device__ inline v16h ld_frag(const _Float16* p) {
  v8h lo = *reinterpret_cast<const v8h*>(p);
  v8h hi = *reinterpret_cast<const v8h*>(p + 16);
  return __builtin_shufflevector(lo, hi, 0,1,2,3,4,5,6,7,8,9,10,11,12,13,14,15);
}

static __device__ inline v8f wmma_f16(v16h a, v16h b, v8f c) {
  return __builtin_amdgcn_wmma_f32_16x16x32_f16(false, a, false, b, (short)0, c,
                                                false, false);
}

// ---------------- prep: x f32 -> f16 ----------------
__global__ void cvt_x_kernel(const float* __restrict__ x, _Float16* __restrict__ xh) {
  int i = (blockIdx.x * blockDim.x + threadIdx.x) * 4;  // 1,048,576 threads * 4
  float4 v = *reinterpret_cast<const float4*>(x + i);
  v4h o;
  o[0] = (_Float16)v.x; o[1] = (_Float16)v.y; o[2] = (_Float16)v.z; o[3] = (_Float16)v.w;
  *reinterpret_cast<v4h*>(xh + i) = o;
}

// ---------------- prep: transpose weights f32 -> f16 (Wt[n][k] = W[k][n]) ----
__global__ void transpose_w_kernel(const float* __restrict__ wq, const float* __restrict__ wk,
                                   const float* __restrict__ wv, const float* __restrict__ wo,
                                   _Float16* __restrict__ wqt, _Float16* __restrict__ wkt,
                                   _Float16* __restrict__ wvt, _Float16* __restrict__ wot) {
  int i = blockIdx.x * blockDim.x + threadIdx.x;       // 4 * 512 * 512 threads
  int mat = i >> 18;
  int rem = i & 262143;
  int n = rem >> 9;
  int k = rem & 511;
  const float* src = (mat == 0) ? wq : (mat == 1) ? wk : (mat == 2) ? wv : wo;
  _Float16* dst    = (mat == 0) ? wqt : (mat == 1) ? wkt : (mat == 2) ? wvt : wot;
  dst[(n << 9) | k] = (_Float16)src[(k << 9) | n];
}

// ---------------- fused QKV projection -------------------------------------
// One wave computes a 32x32 output tile (2x2 WMMA accumulators) of Q, K or V:
// 2 A-frags + 2 B-frags feed 4 WMMAs per k-step (2x operand reuse).
// Q,K stored [b][h][n][d]; V stored transposed [b][h][d][n].
__global__ void qkv_proj_kernel(const _Float16* __restrict__ xh,
                                const _Float16* __restrict__ wqt,
                                const _Float16* __restrict__ wkt,
                                const _Float16* __restrict__ wvt,
                                const float* __restrict__ bq,
                                const float* __restrict__ bk,
                                const float* __restrict__ bv,
                                _Float16* __restrict__ qh,
                                _Float16* __restrict__ kh,
                                _Float16* __restrict__ vt) {
  int wave  = (blockIdx.x * blockDim.x + threadIdx.x) >> 5;
  int lane  = threadIdx.x & 31;
  int col   = wave & 15;          // 16 column tiles of 32 over EMB
  int row   = (wave >> 4) & 255;  // 256 row tiles of 32 over BN
  int which = wave >> 12;         // 0=Q 1=K 2=V
  int g = lane >> 4;
  int m = lane & 15;

  const _Float16* wt   = (which == 0) ? wqt : (which == 1) ? wkt : wvt;
  const float*    bias = (which == 0) ? bq  : (which == 1) ? bk  : bv;

  const _Float16* arow0 = xh + (size_t)(row * 32 + m) * EMB;
  const _Float16* arow1 = xh + (size_t)(row * 32 + 16 + m) * EMB;
  const _Float16* brow0 = wt + (size_t)(col * 32 + m) * EMB;
  const _Float16* brow1 = wt + (size_t)(col * 32 + 16 + m) * EMB;

  v8f acc[2][2] = {};
#pragma unroll
  for (int ks = 0; ks < EMB / 32; ++ks) {
    int o = ks * 32 + g * 8;
    v16h a0 = ld_frag(arow0 + o);
    v16h a1 = ld_frag(arow1 + o);
    v16h b0 = ld_frag(brow0 + o);
    v16h b1 = ld_frag(brow1 + o);
    acc[0][0] = wmma_f16(a0, b0, acc[0][0]);
    acc[0][1] = wmma_f16(a0, b1, acc[0][1]);
    acc[1][0] = wmma_f16(a1, b0, acc[1][0]);
    acc[1][1] = wmma_f16(a1, b1, acc[1][1]);
  }

#pragma unroll
  for (int u = 0; u < 2; ++u) {
#pragma unroll
    for (int t = 0; t < 2; ++t) {
      int gcol = col * 32 + t * 16 + m;   // column this lane owns (C-layout)
      int head = gcol >> 6;
      int doff = gcol & 63;
      int tok0 = row * 32 + u * 16 + 8 * g;  // rows = tok0 + r
      int bi   = tok0 >> 11;
      int nn0  = tok0 & 2047;
      float bval = bias[gcol];
      float vals[8];
#pragma unroll
      for (int r = 0; r < 8; ++r) {
        float v = acc[u][t][r] + bval;
        vals[r] = (which == 0) ? v * QSCALE : v;  // fold softmax scale into Q
      }
      if (which == 2) {
        // V transposed: rows r are consecutive tokens -> one 16B store
        v8h pv;
#pragma unroll
        for (int r = 0; r < 8; ++r) pv[r] = (_Float16)vals[r];
        size_t base = ((size_t)(bi * HEADS + head) * DHEAD + doff) * NSEQ + nn0;
        *reinterpret_cast<v8h*>(vt + base) = pv;
      } else {
        _Float16* dst = (which == 0) ? qh : kh;
        size_t base = ((size_t)(bi * HEADS + head) * NSEQ + nn0) * DHEAD + doff;
#pragma unroll
        for (int r = 0; r < 8; ++r) dst[base + (size_t)r * DHEAD] = (_Float16)vals[r];
      }
    }
  }
}

// ---------------- flash attention ------------------------------------------
// 8 waves per block share one (b,h); waves own consecutive 16-query tiles.
// K/V tiles (32 kv positions) are staged into double-buffered LDS with
// global_load_async_to_lds_b128 (ASYNCcnt) by all 256 threads cooperatively.
__global__ void attn_kernel(const _Float16* __restrict__ qh,
                            const _Float16* __restrict__ kh,
                            const _Float16* __restrict__ vt,
                            _Float16* __restrict__ oh) {
  __shared__ __align__(16) _Float16 kbuf[2][32][72];  // [token][d], padded rows
  __shared__ __align__(16) _Float16 vbuf[2][64][40];  // [d][token], padded rows
  __shared__ __align__(16) _Float16 pbuf[8][16][40];  // P re-layout per wave

  int tid  = threadIdx.x;
  int widx = tid >> 5;
  int lane = tid & 31;
  int g = lane >> 4;
  int m = lane & 15;
  int bh = blockIdx.x >> 4;            // 32 (b,h) pairs
  int bi = bh >> 3;
  int hd = bh & 7;
  int qt = (blockIdx.x & 15) * 8 + widx;

  const _Float16* qp = qh + ((size_t)(bi * HEADS + hd) * NSEQ + qt * 16) * DHEAD;
  const _Float16* kp = kh + (size_t)(bi * HEADS + hd) * NSEQ * DHEAD;
  const _Float16* vp = vt + (size_t)(bi * HEADS + hd) * DHEAD * NSEQ;

  // cooperative staging: each thread moves one 16B chunk of K and one of V
  int ktok = tid >> 3;             // 0..31
  int koff = (tid & 7) * 8;        // halfs within 64-half K row
  int vd   = tid >> 2;             // 0..63
  int voff = (tid & 3) * 8;        // halfs within 32-half V row
  const _Float16* kgbase = kp + (size_t)ktok * DHEAD + koff;
  const _Float16* vgbase = vp + (size_t)vd * NSEQ + voff;

  auto issue = [&](int buf, int kv) {
    uint32_t klds = (uint32_t)(uintptr_t)(&kbuf[buf][ktok][koff]);
    uint64_t kga  = (uint64_t)(uintptr_t)(kgbase + (size_t)kv * 32 * DHEAD);
    asm volatile("global_load_async_to_lds_b128 %0, %1, off"
                 :: "v"(klds), "v"(kga) : "memory");
    uint32_t vlds = (uint32_t)(uintptr_t)(&vbuf[buf][vd][voff]);
    uint64_t vga  = (uint64_t)(uintptr_t)(vgbase + (size_t)kv * 32);
    asm volatile("global_load_async_to_lds_b128 %0, %1, off"
                 :: "v"(vlds), "v"(vga) : "memory");
  };

  // Q A-fragments, loaded once (D=64 -> 2 k-steps of 32)
  v16h qa0 = ld_frag(qp + m * DHEAD + g * 8);
  v16h qa1 = ld_frag(qp + m * DHEAD + 32 + g * 8);

  v8f o0 = {}, o1 = {}, o2 = {}, o3 = {};
  float mr[8], lr[8];
#pragma unroll
  for (int r = 0; r < 8; ++r) { mr[r] = -3.0e38f; lr[r] = 0.0f; }

  _Float16 (*P)[40] = pbuf[widx];

  issue(0, 0);
  int cur = 0;
  for (int kv = 0; kv < NSEQ / 32; ++kv) {
    asm volatile("s_wait_asynccnt 0" ::: "memory");
    __syncthreads();                       // tile `cur` fully in LDS
    if (kv + 1 < NSEQ / 32) issue(cur ^ 1, kv + 1);

    // S = Q * K^T from LDS K tile
    v16h b00 = ld_frag(&kbuf[cur][m][g * 8]);
    v16h b01 = ld_frag(&kbuf[cur][m][32 + g * 8]);
    v16h b10 = ld_frag(&kbuf[cur][16 + m][g * 8]);
    v16h b11 = ld_frag(&kbuf[cur][16 + m][32 + g * 8]);
    v8f s0 = {}, s1 = {};
    s0 = wmma_f16(qa0, b00, s0);
    s0 = wmma_f16(qa1, b01, s0);
    s1 = wmma_f16(qa0, b10, s1);
    s1 = wmma_f16(qa1, b11, s1);

    // online softmax; row (r + 8g) lives in 16 lanes of half-wave g, VGPR r
#pragma unroll
    for (int r = 0; r < 8; ++r) {
      float rm = fmaxf(s0[r], s1[r]);
      rm = fmaxf(rm, __shfl_xor(rm, 1, 32));
      rm = fmaxf(rm, __shfl_xor(rm, 2, 32));
      rm = fmaxf(rm, __shfl_xor(rm, 4, 32));
      rm = fmaxf(rm, __shfl_xor(rm, 8, 32));
      float mnew = fmaxf(mr[r], rm);
      float corr = __expf(mr[r] - mnew);
      float p0 = __expf(s0[r] - mnew);
      float p1 = __expf(s1[r] - mnew);
      float rs = p0 + p1;
      rs += __shfl_xor(rs, 1, 32);
      rs += __shfl_xor(rs, 2, 32);
      rs += __shfl_xor(rs, 4, 32);
      rs += __shfl_xor(rs, 8, 32);
      lr[r] = lr[r] * corr + rs;
      mr[r] = mnew;
      o0[r] *= corr; o1[r] *= corr; o2[r] *= corr; o3[r] *= corr;
      P[r + 8 * g][m]      = (_Float16)p0;   // stage P (C layout) into LDS
      P[r + 8 * g][16 + m] = (_Float16)p1;
    }
    asm volatile("s_wait_dscnt 0" ::: "memory");
    // read back as A-fragment (16x32)
    v8h plo = *reinterpret_cast<const v8h*>(&P[m][g * 8]);
    v8h phi = *reinterpret_cast<const v8h*>(&P[m][16 + g * 8]);
    v16h pa = __builtin_shufflevector(plo, phi,
                                      0,1,2,3,4,5,6,7,8,9,10,11,12,13,14,15);
    // O += P * V from LDS V tile (transposed -> contiguous B-fragments)
    v16h vb0 = ld_frag(&vbuf[cur][0 * 16 + m][g * 8]);
    v16h vb1 = ld_frag(&vbuf[cur][1 * 16 + m][g * 8]);
    v16h vb2 = ld_frag(&vbuf[cur][2 * 16 + m][g * 8]);
    v16h vb3 = ld_frag(&vbuf[cur][3 * 16 + m][g * 8]);
    o0 = wmma_f16(pa, vb0, o0);
    o1 = wmma_f16(pa, vb1, o1);
    o2 = wmma_f16(pa, vb2, o2);
    o3 = wmma_f16(pa, vb3, o3);

    __syncthreads();                       // all reads of `cur` done before reuse
    cur ^= 1;
  }

  // normalize and scatter to [b][n][e] fp16 (heads re-interleaved)
  size_t rowbase = (size_t)(bi * NSEQ + qt * 16 + 8 * g) * EMB + hd * DHEAD + m;
#pragma unroll
  for (int r = 0; r < 8; ++r) {
    float inv = 1.0f / lr[r];
    size_t a = rowbase + (size_t)r * EMB;
    oh[a +  0] = (_Float16)(o0[r] * inv);
    oh[a + 16] = (_Float16)(o1[r] * inv);
    oh[a + 32] = (_Float16)(o2[r] * inv);
    oh[a + 48] = (_Float16)(o3[r] * inv);
  }
}

// ---------------- output projection (f32 out), 32x32 tile per wave ---------
__global__ void out_proj_kernel(const _Float16* __restrict__ oh,
                                const _Float16* __restrict__ wot,
                                const float* __restrict__ bo,
                                float* __restrict__ out) {
  int wave = (blockIdx.x * blockDim.x + threadIdx.x) >> 5;
  int lane = threadIdx.x & 31;
  int col = wave & 15;          // 16 col tiles of 32
  int row = wave >> 4;          // 256 row tiles of 32
  int g = lane >> 4;
  int m = lane & 15;

  const _Float16* arow0 = oh + (size_t)(row * 32 + m) * EMB;
  const _Float16* arow1 = oh + (size_t)(row * 32 + 16 + m) * EMB;
  const _Float16* brow0 = wot + (size_t)(col * 32 + m) * EMB;
  const _Float16* brow1 = wot + (size_t)(col * 32 + 16 + m) * EMB;

  v8f acc[2][2] = {};
#pragma unroll
  for (int ks = 0; ks < EMB / 32; ++ks) {
    int o = ks * 32 + g * 8;
    v16h a0 = ld_frag(arow0 + o);
    v16h a1 = ld_frag(arow1 + o);
    v16h b0 = ld_frag(brow0 + o);
    v16h b1 = ld_frag(brow1 + o);
    acc[0][0] = wmma_f16(a0, b0, acc[0][0]);
    acc[0][1] = wmma_f16(a0, b1, acc[0][1]);
    acc[1][0] = wmma_f16(a1, b0, acc[1][0]);
    acc[1][1] = wmma_f16(a1, b1, acc[1][1]);
  }

#pragma unroll
  for (int u = 0; u < 2; ++u) {
#pragma unroll
    for (int t = 0; t < 2; ++t) {
      int gcol = col * 32 + t * 16 + m;
      float bval = bo[gcol];
      size_t base = (size_t)(row * 32 + u * 16 + 8 * g) * EMB + gcol;
#pragma unroll
      for (int r = 0; r < 8; ++r) out[base + (size_t)r * EMB] = acc[u][t][r] + bval;
    }
  }
}

// ---------------- host launch ----------------------------------------------
extern "C" void kernel_launch(void* const* d_in, const int* in_sizes, int n_in,
                              void* d_out, int out_size, void* d_ws, size_t ws_size,
                              hipStream_t stream) {
  const float* x  = (const float*)d_in[0];
  const float* Wq = (const float*)d_in[1];
  const float* bq = (const float*)d_in[2];
  const float* Wk = (const float*)d_in[3];
  const float* bk = (const float*)d_in[4];
  const float* Wv = (const float*)d_in[5];
  const float* bv = (const float*)d_in[6];
  const float* Wo = (const float*)d_in[7];
  const float* bo = (const float*)d_in[8];

  char* ws = (char*)d_ws;
  size_t off = 0;
  _Float16* xh  = (_Float16*)(ws + off); off += (size_t)BN * EMB * 2;
  _Float16* wqt = (_Float16*)(ws + off); off += (size_t)EMB * EMB * 2;
  _Float16* wkt = (_Float16*)(ws + off); off += (size_t)EMB * EMB * 2;
  _Float16* wvt = (_Float16*)(ws + off); off += (size_t)EMB * EMB * 2;
  _Float16* wot = (_Float16*)(ws + off); off += (size_t)EMB * EMB * 2;
  _Float16* qh  = (_Float16*)(ws + off); off += (size_t)BN * EMB * 2;
  _Float16* kh  = (_Float16*)(ws + off); off += (size_t)BN * EMB * 2;
  _Float16* vt  = (_Float16*)(ws + off); off += (size_t)BN * EMB * 2;
  _Float16* oh  = (_Float16*)(ws + off); off += (size_t)BN * EMB * 2;
  (void)ws_size; (void)in_sizes; (void)n_in; (void)out_size;

  // 1) x -> fp16
  cvt_x_kernel<<<4096, 256, 0, stream>>>(x, xh);
  // 2) weights -> transposed fp16
  transpose_w_kernel<<<4096, 256, 0, stream>>>(Wq, Wk, Wv, Wo, wqt, wkt, wvt, wot);
  // 3) QKV projection: 3 * 256 * 16 wave-tiles of 32x32, 8 waves/block
  qkv_proj_kernel<<<1536, 256, 0, stream>>>(xh, wqt, wkt, wvt, bq, bk, bv, qh, kh, vt);
  // 4) flash attention: 32 (b,h) * 16 blocks, 8 query-tiles per block
  attn_kernel<<<512, 256, 0, stream>>>(qh, kh, vt, oh);
  // 5) output projection: 256 * 16 wave-tiles of 32x32
  out_proj_kernel<<<512, 256, 0, stream>>>(oh, wot, bo, (float*)d_out);
}
